// Predictor_24180665876995
// MI455X (gfx1250) — matrix operand order repaired and checked by reading, not compiled, not run on hardware
//
#include <hip/hip_runtime.h>
#include <hip/hip_bf16.h>
#include <math.h>

// Problem constants (match reference)
#define B_SETS 64
#define L_MAX  1024
#define TOTAL  (B_SETS * L_MAX)   // 65536
#define DMODEL 512
#define HDIM   2048
#define NHEADS 8
#define HEADD  64
#define NEGF   (-1e30f)

typedef __attribute__((ext_vector_type(16))) __bf16 v16bf;
typedef __attribute__((ext_vector_type(8)))  __bf16 v8bf;
typedef __attribute__((ext_vector_type(8)))  float  v8f;

// Async memory->LDS copy of 16 bytes per lane (CDNA5 VGLOBAL async op,
// tracked by ASYNCcnt). lds_off is the addrspace(3) byte offset (low 32 bits
// of the generic address), gaddr the 64-bit global address.
__device__ __forceinline__ void async_copy_b128(unsigned lds_off,
                                                unsigned long long gaddr) {
  asm volatile("global_load_async_to_lds_b128 %0, %1, off"
               :: "v"(lds_off), "v"(gaddr)
               : "memory");
}
__device__ __forceinline__ void wait_async0() {
  asm volatile("s_wait_asynccnt 0x0" ::: "memory");
}

// ---------------------------------------------------------------------------
// Weight convert + transpose: Wt[n][k] = bf16(W[k][n])   (Wt is [N,K])
// ---------------------------------------------------------------------------
__global__ void transpose_to_bf16(const float* __restrict__ W,
                                  __bf16* __restrict__ Wt, int K, int N) {
  int idx = blockIdx.x * blockDim.x + threadIdx.x;
  if (idx >= K * N) return;
  int n = idx / K;
  int k = idx - n * K;
  Wt[idx] = (__bf16)W[(size_t)k * N + n];
}

// ---------------------------------------------------------------------------
// qp = q @ Wq + bq   (512 outputs, trivial)
// ---------------------------------------------------------------------------
__global__ void compute_qp(const float* __restrict__ q,
                           const float* __restrict__ Wq,
                           const float* __restrict__ bq,
                           float* __restrict__ qp) {
  int j = blockIdx.x * blockDim.x + threadIdx.x;
  if (j >= DMODEL) return;
  float s = bq[j];
  for (int d = 0; d < DMODEL; ++d) s = fmaf(q[d], Wq[(size_t)d * DMODEL + j], s);
  qp[j] = s;
}

// ---------------------------------------------------------------------------
// CSR gather + fp32->bf16 embed:  z[b*L+l, :] = emb[mut_idx[offsets[b]+l], :]
// Padded (l >= len) rows are zero-filled (workspace is poisoned otherwise).
// ---------------------------------------------------------------------------
__global__ __launch_bounds__(128) void gather_embed(
    const int* __restrict__ mut_idx, const int* __restrict__ offsets,
    const float* __restrict__ emb, __bf16* __restrict__ z) {
  int t = blockIdx.x;
  int b = t >> 10;            // / L_MAX
  int l = t & (L_MAX - 1);
  int beg = offsets[b];
  int len = offsets[b + 1] - beg;
  int d = threadIdx.x * 4;
  __bf16* zr = z + (size_t)t * DMODEL + d;
  if (l < len) {
    int idx = mut_idx[beg + l];
    float4 e = *(const float4*)(emb + (size_t)idx * DMODEL + d);
    zr[0] = (__bf16)e.x; zr[1] = (__bf16)e.y;
    zr[2] = (__bf16)e.z; zr[3] = (__bf16)e.w;
  } else {
    zr[0] = (__bf16)0.f; zr[1] = (__bf16)0.f;
    zr[2] = (__bf16)0.f; zr[3] = (__bf16)0.f;
  }
}

// ---------------------------------------------------------------------------
// bf16 WMMA GEMM:  C[M,N] = act( A[M,K] @ Bt[N,K]^T + bias[N] )
// Block tile 128(M) x 128(N), K-step 32, 256 threads = 8 wave32s.
// Wave (wm in 0..3, wn in 0..1) computes a 32x64 strip = 8 WMMA tiles
// (2 A-frags x 4 B-frags -> 8 v_wmma per 12 ds_load_b128).
// Tiles are staged memory->LDS with GLOBAL_LOAD_ASYNC_TO_LDS_B128 into a
// double buffer: issue next K-tile's async copies, compute current, then
// s_wait_asynccnt 0 + one barrier per iteration.
// A/B fragments use contiguous k order k = 16*half + i, the same for A and B.
// ---------------------------------------------------------------------------
__global__ __launch_bounds__(256) void gemm_bf16_wmma(
    const __bf16* __restrict__ A, const __bf16* __restrict__ Bt,
    const float* __restrict__ bias, __bf16* __restrict__ C,
    int M, int N, int K, int relu) {
  __shared__ __align__(64) __bf16 ldsA[2][128][32];   // 2 x 8 KB
  __shared__ __align__(64) __bf16 ldsB[2][128][32];   // 2 x 8 KB (Bt rows [n][k])

  const int tid    = threadIdx.x;
  const int lane   = tid & 31;
  const int wid    = tid >> 5;
  const int wave_m = wid & 3;    // 0..3  -> M strip (32 rows each)
  const int wave_n = wid >> 2;   // 0..1  -> N strip (64 cols each)
  const int half   = lane >> 4;  // selects K/M halves per ISA layout
  const int nrow   = lane & 15;  // A: row m, B/C/D: column n
  const int m0 = blockIdx.y * 128;
  const int n0 = blockIdx.x * 128;

  // Staging assignment: 128x32 bf16 tile = 512 x 16B chunks, 2 per thread.
  const int srow0 = tid >> 1;           // chunk pair base: rows 0..127
  // (use flat mapping below instead)

  v8f acc[2][4];
#pragma unroll
  for (int i = 0; i < 2; ++i)
#pragma unroll
    for (int t = 0; t < 4; ++t) acc[i][t] = (v8f){};

  // ---- stage first K-tile (buffer 0) ----
#pragma unroll
  for (int v = 0; v < 2; ++v) {
    int flat = tid + v * 256;           // 0..511
    int row = flat >> 2, colv = flat & 3;
    async_copy_b128((unsigned)(size_t)&ldsA[0][row][colv * 8],
                    (unsigned long long)(size_t)(A + (size_t)(m0 + row) * K + colv * 8));
    async_copy_b128((unsigned)(size_t)&ldsB[0][row][colv * 8],
                    (unsigned long long)(size_t)(Bt + (size_t)(n0 + row) * K + colv * 8));
  }
  wait_async0();
  __syncthreads();

  for (int k0 = 0; k0 < K; k0 += 32) {
    const int buf = (k0 >> 5) & 1;
    const bool more = (k0 + 32) < K;

    // ---- issue async copies for the next K-tile into the other buffer ----
    if (more) {
      const int nk = k0 + 32;
#pragma unroll
      for (int v = 0; v < 2; ++v) {
        int flat = tid + v * 256;
        int row = flat >> 2, colv = flat & 3;
        async_copy_b128((unsigned)(size_t)&ldsA[buf ^ 1][row][colv * 8],
                        (unsigned long long)(size_t)(A + (size_t)(m0 + row) * K + nk + colv * 8));
        async_copy_b128((unsigned)(size_t)&ldsB[buf ^ 1][row][colv * 8],
                        (unsigned long long)(size_t)(Bt + (size_t)(n0 + row) * K + nk + colv * 8));
      }
    }

    // ---- fragments from current buffer ----
    // A 16x32 frag: lane m=lane&15, half h: a[i]=A[m][16h+i]
    const __bf16* ap0 = &ldsA[buf][wave_m * 32 + nrow][half * 16];
    const __bf16* ap1 = ap0 + 16 * 32;  // +16 rows
    v8bf a0lo = *(const v8bf*)ap0;
    v8bf a0hi = *(const v8bf*)(ap0 + 8);
    v8bf a1lo = *(const v8bf*)ap1;
    v8bf a1hi = *(const v8bf*)(ap1 + 8);
    v16bf afrag0 = __builtin_shufflevector(
        a0lo, a0hi, 0, 1, 2, 3, 4, 5, 6, 7, 8, 9, 10, 11, 12, 13, 14, 15);
    v16bf afrag1 = __builtin_shufflevector(
        a1lo, a1hi, 0, 1, 2, 3, 4, 5, 6, 7, 8, 9, 10, 11, 12, 13, 14, 15);

    // B 32x16 frags: b[i] = B[16h+i][n] = Bt[n][16h+i] -> contiguous 32B
    const __bf16* bbase = &ldsB[buf][wave_n * 64 + nrow][half * 16];
    v16bf bf0 = *(const v16bf*)(bbase + 0 * 512);  // 512 = 16 rows * 32 elems
    v16bf bf1 = *(const v16bf*)(bbase + 1 * 512);
    v16bf bf2 = *(const v16bf*)(bbase + 2 * 512);
    v16bf bf3 = *(const v16bf*)(bbase + 3 * 512);

    acc[0][0] = __builtin_amdgcn_wmma_f32_16x16x32_bf16(false, afrag0, false, bf0,
                                                        (short)0, acc[0][0], false, false);
    acc[0][1] = __builtin_amdgcn_wmma_f32_16x16x32_bf16(false, afrag0, false, bf1,
                                                        (short)0, acc[0][1], false, false);
    acc[0][2] = __builtin_amdgcn_wmma_f32_16x16x32_bf16(false, afrag0, false, bf2,
                                                        (short)0, acc[0][2], false, false);
    acc[0][3] = __builtin_amdgcn_wmma_f32_16x16x32_bf16(false, afrag0, false, bf3,
                                                        (short)0, acc[0][3], false, false);
    acc[1][0] = __builtin_amdgcn_wmma_f32_16x16x32_bf16(false, afrag1, false, bf0,
                                                        (short)0, acc[1][0], false, false);
    acc[1][1] = __builtin_amdgcn_wmma_f32_16x16x32_bf16(false, afrag1, false, bf1,
                                                        (short)0, acc[1][1], false, false);
    acc[1][2] = __builtin_amdgcn_wmma_f32_16x16x32_bf16(false, afrag1, false, bf2,
                                                        (short)0, acc[1][2], false, false);
    acc[1][3] = __builtin_amdgcn_wmma_f32_16x16x32_bf16(false, afrag1, false, bf3,
                                                        (short)0, acc[1][3], false, false);

    // Own async copies must land before anyone signals; reads of `buf` are
    // complete (consumed by WMMA above). One barrier per iteration.
    if (more) wait_async0();
    __syncthreads();
  }
  (void)srow0;

  // Epilogue: D tile layout -> row = 8*half + r, col = lane&15.
  const int colbase = n0 + wave_n * 64 + nrow;
  const int rowbase = m0 + wave_m * 32 + half * 8;
#pragma unroll
  for (int i = 0; i < 2; ++i) {
#pragma unroll
    for (int t = 0; t < 4; ++t) {
      int col = colbase + t * 16;
      float bval = bias[col];
#pragma unroll
      for (int r = 0; r < 8; ++r) {
        float val = acc[i][t][r] + bval;
        if (relu) val = fmaxf(val, 0.f);
        C[(size_t)(rowbase + i * 16 + r) * N + col] = (__bf16)val;
      }
    }
  }
}

// ---------------------------------------------------------------------------
// Attention pool, one block per (set b, head): masked scores -> softmax ->
// weighted sum of V into ctx[b, head*64:head*64+64].
// ---------------------------------------------------------------------------
__global__ __launch_bounds__(256) void attn_pool(
    const __bf16* __restrict__ kbuf, const __bf16* __restrict__ vbuf,
    const float* __restrict__ qp, const int* __restrict__ offsets,
    float* __restrict__ ctx) {
  const int b = blockIdx.x, head = blockIdx.y, tid = threadIdx.x;
  __shared__ float sc[L_MAX];
  __shared__ float red[256];
  __shared__ float part[256];
  const int len = offsets[b + 1] - offsets[b];
  const float scale = 0.125f;  // 1/sqrt(64)

  float lmax = NEGF;
#pragma unroll
  for (int c = 0; c < 4; ++c) {
    int l = tid + c * 256;
    float s = NEGF;
    if (l < len) {
      const __bf16* kr = kbuf + (size_t)(b * L_MAX + l) * DMODEL + head * HEADD;
      float dot = 0.f;
      for (int d = 0; d < HEADD; ++d)
        dot = fmaf(qp[head * HEADD + d], (float)kr[d], dot);
      s = dot * scale;
    }
    sc[l] = s;
    lmax = fmaxf(lmax, s);
  }
  red[tid] = lmax;
  __syncthreads();
  for (int s = 128; s > 0; s >>= 1) {
    if (tid < s) red[tid] = fmaxf(red[tid], red[tid + s]);
    __syncthreads();
  }
  const float mx = red[0];
  __syncthreads();

  float lsum = 0.f;
#pragma unroll
  for (int c = 0; c < 4; ++c) {
    int l = tid + c * 256;
    float e = expf(sc[l] - mx);  // masked lanes underflow to 0
    sc[l] = e;
    lsum += e;
  }
  red[tid] = lsum;
  __syncthreads();
  for (int s = 128; s > 0; s >>= 1) {
    if (tid < s) red[tid] += red[tid + s];
    __syncthreads();
  }
  const float inv = 1.0f / red[0];
  __syncthreads();

  const int hd = tid & 63;
  const int chunk = tid >> 6;  // 4 chunks of 256 keys
  float acc = 0.f;
  for (int l = chunk * 256; l < (chunk + 1) * 256; ++l) {
    acc = fmaf(sc[l],
               (float)vbuf[(size_t)(b * L_MAX + l) * DMODEL + head * HEADD + hd],
               acc);
  }
  part[tid] = acc;
  __syncthreads();
  if (chunk == 0) {
    float total = (part[hd] + part[hd + 64] + part[hd + 128] + part[hd + 192]) * inv;
    ctx[(size_t)b * DMODEL + head * HEADD + hd] = total;
  }
}

// ---------------------------------------------------------------------------
// Head: pooled = ctx@Wo+bo, LayerNorm, relu(y@Wh1+bh1)@Wh2+bh2 -> logit[b]
// ---------------------------------------------------------------------------
__global__ __launch_bounds__(256) void head_kernel(
    const float* __restrict__ ctx, const float* __restrict__ Wo,
    const float* __restrict__ bo, const float* __restrict__ ln_g,
    const float* __restrict__ ln_b, const float* __restrict__ Wh1,
    const float* __restrict__ bh1, const float* __restrict__ Wh2,
    const float* __restrict__ bh2, float* __restrict__ out) {
  const int b = blockIdx.x, tid = threadIdx.x;
  __shared__ float pooled[DMODEL];
  __shared__ float y[DMODEL];
  __shared__ float red[256];
  __shared__ float s_mu, s_rstd;
  const float* cb = ctx + (size_t)b * DMODEL;

  for (int j = tid; j < DMODEL; j += 256) {
    float s = bo[j];
    for (int d = 0; d < DMODEL; ++d) s = fmaf(cb[d], Wo[(size_t)d * DMODEL + j], s);
    pooled[j] = s;
  }
  __syncthreads();

  red[tid] = pooled[tid] + pooled[tid + 256];
  __syncthreads();
  for (int s = 128; s > 0; s >>= 1) {
    if (tid < s) red[tid] += red[tid + s];
    __syncthreads();
  }
  if (tid == 0) s_mu = red[0] * (1.0f / DMODEL);
  __syncthreads();
  const float mu = s_mu;

  float d0 = pooled[tid] - mu, d1 = pooled[tid + 256] - mu;
  red[tid] = d0 * d0 + d1 * d1;
  __syncthreads();
  for (int s = 128; s > 0; s >>= 1) {
    if (tid < s) red[tid] += red[tid + s];
    __syncthreads();
  }
  if (tid == 0) s_rstd = rsqrtf(red[0] * (1.0f / DMODEL) + 1e-5f);
  __syncthreads();
  const float rstd = s_rstd;

  for (int j = tid; j < DMODEL; j += 256)
    y[j] = (pooled[j] - mu) * rstd * ln_g[j] + ln_b[j];
  __syncthreads();

  float p = 0.f;
  for (int j = tid; j < HDIM; j += 256) {
    float s = bh1[j];
    for (int d = 0; d < DMODEL; ++d) s = fmaf(y[d], Wh1[(size_t)d * HDIM + j], s);
    p += fmaxf(s, 0.f) * Wh2[j];
  }
  red[tid] = p;
  __syncthreads();
  for (int s = 128; s > 0; s >>= 1) {
    if (tid < s) red[tid] += red[tid + s];
    __syncthreads();
  }
  if (tid == 0) out[b] = red[0] + bh2[0];
}

// ---------------------------------------------------------------------------
// Launcher. Workspace (~330 MB): bf16 weights, z (reused for h), act (reused
// for k|v), qp, ctx.
// ---------------------------------------------------------------------------
extern "C" void kernel_launch(void* const* d_in, const int* in_sizes, int n_in,
                              void* d_out, int out_size, void* d_ws,
                              size_t ws_size, hipStream_t stream) {
  (void)in_sizes; (void)n_in; (void)out_size; (void)ws_size;
  const int*   mut_idx = (const int*)d_in[0];
  const int*   offsets = (const int*)d_in[1];
  const float* emb  = (const float*)d_in[2];
  const float* W1   = (const float*)d_in[3];
  const float* b1   = (const float*)d_in[4];
  const float* W2   = (const float*)d_in[5];
  const float* b2   = (const float*)d_in[6];
  const float* q    = (const float*)d_in[7];
  const float* Wq   = (const float*)d_in[8];
  const float* bq   = (const float*)d_in[9];
  const float* Wk   = (const float*)d_in[10];
  const float* bk   = (const float*)d_in[11];
  const float* Wv   = (const float*)d_in[12];
  const float* bv   = (const float*)d_in[13];
  const float* Wo   = (const float*)d_in[14];
  const float* bo   = (const float*)d_in[15];
  const float* ln_g = (const float*)d_in[16];
  const float* ln_b = (const float*)d_in[17];
  const float* Wh1  = (const float*)d_in[18];
  const float* bh1  = (const float*)d_in[19];
  const float* Wh2  = (const float*)d_in[20];
  const float* bh2  = (const float*)d_in[21];
  float* out = (float*)d_out;

  char* ws = (char*)d_ws;
  size_t off = 0;
  auto carve = [&](size_t bytes) -> char* {
    char* p = ws + off;
    off = (off + bytes + 255) & ~(size_t)255;
    return p;
  };
  __bf16* W1t  = (__bf16*)carve((size_t)DMODEL * HDIM * 2);    // [N=H,K=D]
  __bf16* W2t  = (__bf16*)carve((size_t)DMODEL * HDIM * 2);    // [N=D,K=H]
  __bf16* Wkt  = (__bf16*)carve((size_t)DMODEL * DMODEL * 2);
  __bf16* Wvt  = (__bf16*)carve((size_t)DMODEL * DMODEL * 2);
  float*  qp   = (float*)carve(DMODEL * 4);
  float*  ctx  = (float*)carve((size_t)B_SETS * DMODEL * 4);
  __bf16* zbuf = (__bf16*)carve((size_t)TOTAL * DMODEL * 2);   // 64 MB
  __bf16* act  = (__bf16*)carve((size_t)TOTAL * HDIM * 2);     // 256 MB
  __bf16* hbuf = zbuf;                               // alias: z dead after gemm1
  __bf16* kbuf = act;                                // alias: act dead after gemm2
  __bf16* vbuf = act + (size_t)TOTAL * DMODEL;

  // 1) weights -> bf16, transposed to [N,K]
  {
    int n = DMODEL * HDIM;
    transpose_to_bf16<<<(n + 255) / 256, 256, 0, stream>>>(W1, W1t, DMODEL, HDIM);
    transpose_to_bf16<<<(n + 255) / 256, 256, 0, stream>>>(W2, W2t, HDIM, DMODEL);
    int n2 = DMODEL * DMODEL;
    transpose_to_bf16<<<(n2 + 255) / 256, 256, 0, stream>>>(Wk, Wkt, DMODEL, DMODEL);
    transpose_to_bf16<<<(n2 + 255) / 256, 256, 0, stream>>>(Wv, Wvt, DMODEL, DMODEL);
  }
  // 2) qp = q@Wq + bq
  compute_qp<<<2, 256, 0, stream>>>(q, Wq, bq, qp);
  // 3) gather + bf16 convert
  gather_embed<<<TOTAL, 128, 0, stream>>>(mut_idx, offsets, emb, zbuf);
  // 4) phi MLP (274-GFLOP hot path) + K/V projections, all WMMA
  dim3 g1(HDIM / 128, TOTAL / 128);
  gemm_bf16_wmma<<<g1, 256, 0, stream>>>(zbuf, W1t, b1, act, TOTAL, HDIM, DMODEL, 1);
  dim3 g2(DMODEL / 128, TOTAL / 128);
  gemm_bf16_wmma<<<g2, 256, 0, stream>>>(act, W2t, b2, hbuf, TOTAL, DMODEL, HDIM, 0);
  gemm_bf16_wmma<<<g2, 256, 0, stream>>>(hbuf, Wkt, bk, kbuf, TOTAL, DMODEL, DMODEL, 0);
  gemm_bf16_wmma<<<g2, 256, 0, stream>>>(hbuf, Wvt, bv, vbuf, TOTAL, DMODEL, DMODEL, 0);
  // 5) attention pool + head
  dim3 ga(B_SETS, NHEADS);
  attn_pool<<<ga, 256, 0, stream>>>(kbuf, vbuf, qp, offsets, ctx);
  head_kernel<<<B_SETS, 256, 0, stream>>>(ctx, Wo, bo, ln_g, ln_b, Wh1, bh1, Wh2,
                                          bh2, out);
}
